// DailyCrossSourceFusion_47974784697035
// MI455X (gfx1250) — matrix operand order repaired and checked by reading, not compiled, not run on hardware
//
#include <hip/hip_runtime.h>
#include <hip/hip_bf16.h>
#include <math.h>

typedef __bf16 bf16;
typedef __attribute__((ext_vector_type(16))) __bf16 v16bf;
typedef __attribute__((ext_vector_type(8)))  __bf16 v8bf_t;
typedef __attribute__((ext_vector_type(8)))  float   v8f;

#define DM 256
#define NSRC 5
#define NSEQ_TOTAL 16384
#define SEQ_PER_BLK 8
#define TOK 40            // real tokens per block (8 seq * 5)
#define MPAD 48           // padded to 3 M-tiles of 16
#define LDX 272           // bf16 element stride for x / attnout (mult of 16)
#define HQ 112            // per-head qkv buffer stride
#define HF 80             // ffn hidden-chunk stride
#define H2LD 528          // projection hidden stride (bf16)
#define GLD 264           // f32 scratch stride

// shared memory carve (bytes)
#define XOFF 0
#define SOFF 26112        // 48*272*2
#define HOFF 52224        // + 48*272*2
#define IOFF 62976        // + 48*112*2
#define SMEM_BYTES 63168  // + 40*4 + pad

// gfx1250 async global->LDS path (guarded: falls back if builtins absent)
#if defined(__has_builtin)
#if __has_builtin(__builtin_amdgcn_global_load_async_to_lds_b128) && \
    __has_builtin(__builtin_amdgcn_s_wait_asynccnt)
#define HAVE_ASYNC_LDS 1
#endif
#endif
#ifndef HAVE_ASYNC_LDS
#define HAVE_ASYNC_LDS 0
#endif

#if HAVE_ASYNC_LDS
#define AS1 __attribute__((address_space(1)))
#define AS3 __attribute__((address_space(3)))
typedef int v4i_t __attribute__((vector_size(16)));
#endif

// ---------------- device helpers ----------------

__device__ __forceinline__ float wred(float v) {
#pragma unroll
  for (int m = 16; m > 0; m >>= 1) v += __shfl_xor(v, m, 32);
  return v;
}

__device__ __forceinline__ float gelu_f(float x) {
  return 0.5f * x * (1.0f + erff(x * 0.70710678118654752f));
}

__device__ __forceinline__ v8f vzero8() {
  v8f z = {0.f, 0.f, 0.f, 0.f, 0.f, 0.f, 0.f, 0.f};
  return z;
}

// A-fragment (16x32 bf16) from LDS, row-major stride lda.
// CDNA5 layout: lanes 0-15 M=lane, K={k0..k0+7, k0+16..k0+23};
//               lanes 16-31 M=lane-16, K={k0+8..k0+15, k0+24..k0+31}.
__device__ __forceinline__ v16bf load_a_lds(const bf16* A, int lda, int row0, int k0) {
  int lane = threadIdx.x & 31;
  const bf16* p = A + (row0 + (lane & 15)) * lda + k0 + ((lane >> 4) << 3);
  v8bf_t lo = *(const v8bf_t*)p;
  v8bf_t hi = *(const v8bf_t*)(p + 16);
  v16bf a;
#pragma unroll
  for (int i = 0; i < 8; i++) { a[i] = lo[i]; a[i + 8] = hi[i]; }
  return a;
}

// B-fragment (32x16 bf16) streamed from global weights W[N][K] row-major.
// lanes 0-15: N=lane, K={k0..k0+7, k0+16..k0+23}; lanes 16-31: N=lane-16, K={k0+8..k0+15, k0+24..k0+31}.
__device__ __forceinline__ v16bf load_b_glb(const bf16* W, int ldw, int n0, int k0) {
  int lane = threadIdx.x & 31;
  const bf16* p = W + (size_t)(n0 + (lane & 15)) * ldw + k0 + ((lane >> 4) << 3);
  v8bf_t lo = *(const v8bf_t*)p;
  v8bf_t hi = *(const v8bf_t*)(p + 16);
  v16bf b;
#pragma unroll
  for (int i = 0; i < 8; i++) { b[i] = lo[i]; b[i + 8] = hi[i]; }
  return b;
}

// A-fragment for the "concatenated sources" view: row = sequence (8 real),
// K spans 1280 = 5 tokens * 256; each 32-wide K-step stays in one token row.
__device__ __forceinline__ v16bf load_a_cc(const bf16* xb, int k0) {
  int lane = threadIdx.x & 31;
  int s = lane & 15;
  v16bf a;
  if (s >= SEQ_PER_BLK) {
#pragma unroll
    for (int i = 0; i < 16; i++) a[i] = (bf16)0.f;
    return a;
  }
  int kb = k0 + ((lane >> 4) << 3);
  int tok = s * 5 + (k0 >> 8);
  int col = kb & 255;
  const bf16* p = xb + tok * LDX + col;
  v8bf_t lo = *(const v8bf_t*)p;
  v8bf_t hi = *(const v8bf_t*)(p + 16);
#pragma unroll
  for (int i = 0; i < 8; i++) { a[i] = lo[i]; a[i + 8] = hi[i]; }
  return a;
}

__device__ __forceinline__ v8f wmma_k(const bf16* A, int lda, int row0,
                                      const bf16* W, int ldw, int n0,
                                      int K, v8f acc) {
  for (int k0 = 0; k0 < K; k0 += 32) {
    v16bf a = load_a_lds(A, lda, row0, k0);
    v16bf b = load_b_glb(W, ldw, n0, k0);
    acc = __builtin_amdgcn_wmma_f32_16x16x32_bf16(false, a, false, b,
                                                  (short)0, acc, false, false);
  }
  return acc;
}

// Store a 16x16 f32 C tile (bias, optional gelu) into LDS bf16 buffer.
// C layout: VGPR r -> lanes 0-15 M=r, lanes 16-31 M=r+8; N = lane&15.
__device__ __forceinline__ void store_tile_bf16_bias(bf16* buf, int ld, int row0, int col0,
                                                     v8f acc, const float* bias, int bias0,
                                                     bool do_gelu) {
  int lane = threadIdx.x & 31;
  int rbase = row0 + ((lane >> 4) << 3);
  int col = col0 + (lane & 15);
  float b = bias[bias0 + (lane & 15)];
#pragma unroll
  for (int r = 0; r < 8; r++) {
    float v = acc[r] + b;
    if (do_gelu) v = gelu_f(v);
    buf[(rbase + r) * ld + col] = (bf16)v;
  }
}

// LayerNorm rows in place (bf16 buffer), one wave per row, optional gelu after.
__device__ __forceinline__ void ln_rows_bf16(bf16* buf, int ld, int nrows, int W,
                                             const float* g, const float* b, bool do_gelu) {
  int wave = threadIdx.x >> 5, lane = threadIdx.x & 31;
  for (int r = wave; r < nrows; r += 8) {
    bf16* row = buf + r * ld;
    float s = 0.f, s2 = 0.f;
    for (int d = lane; d < W; d += 32) { float v = (float)row[d]; s += v; s2 += v * v; }
    s = wred(s); s2 = wred(s2);
    float m = s / W, var = s2 / W - m * m;
    float inv = rsqrtf(var + 1e-5f);
    for (int d = lane; d < W; d += 32) {
      float v = ((float)row[d] - m) * inv * g[d] + b[d];
      if (do_gelu) v = gelu_f(v);
      row[d] = (bf16)v;
    }
  }
}

// ---------------- params ----------------

struct Params {
  const float* h0; const float* h1; const float* h2s; const float* h3; const float* h4;
  const float* type_emb;
  const bf16* aw[2]; const float* ab[2];
  const bf16* ow[2]; const float* ob[2];
  const float* lg[2]; const float* lb[2];
  const bf16* fw1[2]; const float* fb1[2];
  const bf16* fw2[2]; const float* fb2[2];
  const float* ng[2]; const float* nb[2];
  const bf16* gw1; const float* gb1; const float* glg; const float* glb;
  const float* gw2; const float* gb2;
  const bf16* pw1; const float* pb1; const float* pl1g; const float* pl1b;
  const bf16* pw2; const float* pb2; const float* pl2g; const float* pl2b;
  float* out;
};

__device__ __forceinline__ const float* src_ptr(const Params& p, int l) {
  const float* r = p.h0;
  if (l == 1) r = p.h1;
  else if (l == 2) r = p.h2s;
  else if (l == 3) r = p.h3;
  else if (l == 4) r = p.h4;
  return r;
}

// ---------------- main fused kernel ----------------

__global__ __launch_bounds__(256) void fusion_kernel(Params p) {
  __shared__ __align__(32) unsigned char smem[SMEM_BYTES];
  bf16* xb = (bf16*)(smem + XOFF);       // x / residual, [48][272]
  bf16* sb = (bf16*)(smem + SOFF);       // attnout, later proj-hidden h2
  bf16* hb = (bf16*)(smem + HOFF);       // per-head qkv / ffn hidden chunk / f32 stage
  float* imp = (float*)(smem + IOFF);    // gating weights [40]
  float* gf = (float*)(smem + SOFF);     // f32 view of sb (gating pre-LN)
  bf16* h2 = (bf16*)(smem + SOFF);       // bf16 view of sb (proj hidden)
  float* yf = (float*)(smem + XOFF);     // f32 view of xb (final pre-LN)

  const int tid = threadIdx.x;
  const int wave = tid >> 5;
  const int lane = tid & 31;
  const int base = blockIdx.x * SEQ_PER_BLK;

  // Warm L2 for the next block's inputs (lowers to global_prefetch_b8).
  if (base + SEQ_PER_BLK < NSEQ_TOTAL && tid < 64) {
#pragma unroll
    for (int l = 0; l < NSRC; l++) {
      const char* np = (const char*)(src_ptr(p, l) + (size_t)(base + SEQ_PER_BLK) * DM);
      __builtin_prefetch(np + tid * 128, 0, 0);
    }
  }

  // ---- Phase 0: build x = stack(sources) + type_emb, cast bf16 into LDS ----
#if HAVE_ASYNC_LDS
  {
    float* stage = (float*)hb;           // 8KB f32 staging per source
    for (int l = 0; l < NSRC; l++) {
      const float* srcp = src_ptr(p, l) + (size_t)base * DM;
      for (int i = tid; i < (SEQ_PER_BLK * DM) / 4; i += 256) {   // 512 x 16B
        __builtin_amdgcn_global_load_async_to_lds_b128(
            (AS1 v4i_t*)(srcp + i * 4), (AS3 v4i_t*)(stage + i * 4), 0, 0);
      }
      __builtin_amdgcn_s_wait_asynccnt(0);
      __syncthreads();
      for (int idx = tid; idx < SEQ_PER_BLK * DM; idx += 256) {
        int seq = idx >> 8, d = idx & 255;
        xb[(seq * 5 + l) * LDX + d] = (bf16)(stage[idx] + p.type_emb[l * DM + d]);
      }
      __syncthreads();
    }
  }
#else
  for (int idx = tid; idx < TOK * DM; idx += 256) {
    int tok = idx >> 8, d = idx & 255;
    int seq = tok / 5, l = tok % 5;
    float v = src_ptr(p, l)[(size_t)(base + seq) * DM + d] + p.type_emb[l * DM + d];
    xb[tok * LDX + d] = (bf16)v;
  }
#endif
  for (int idx = tid; idx < (MPAD - TOK) * DM; idx += 256) {
    int tok = TOK + (idx >> 8), d = idx & 255;
    xb[tok * LDX + d] = (bf16)0.f;
    sb[tok * LDX + d] = (bf16)0.f;
  }
  __syncthreads();

  // ---- Two transformer layers ----
  for (int layer = 0; layer < 2; layer++) {
    const bf16* AW = p.aw[layer]; const float* AB = p.ab[layer];

    // Phase A: per-head QKV (WMMA) + tiny 5x5 attention (VALU)
    for (int head = 0; head < 8; head++) {
      for (int t = wave; t < 18; t += 8) {       // 3 M-tiles x 6 N-tiles (q,k,v x 2)
        int mt = t / 6, nt = t % 6;
        int sec = nt >> 1, half = nt & 1;
        int n0 = sec * 256 + head * 32 + half * 16;   // row in packed aw [768][256]
        v8f acc = wmma_k(xb, LDX, mt * 16, AW, 256, n0, 256, vzero8());
        store_tile_bf16_bias(hb, HQ, mt * 16, sec * 32 + half * 16, acc, AB, n0, false);
      }
      __syncthreads();
      if (tid < TOK) {
        int seq = tid / 5, qi = tid % 5;
        const bf16* qrow = hb + (seq * 5 + qi) * HQ;
        float srow[5];
        float mx = -1e30f;
#pragma unroll
        for (int j = 0; j < 5; j++) {
          const bf16* krow = hb + (seq * 5 + j) * HQ + 32;
          float a = 0.f;
          for (int d = 0; d < 32; d++) a += (float)qrow[d] * (float)krow[d];
          srow[j] = a * 0.17677669529663687f;   // 1/sqrt(32)
          mx = fmaxf(mx, srow[j]);
        }
        float den = 0.f;
#pragma unroll
        for (int j = 0; j < 5; j++) { srow[j] = __expf(srow[j] - mx); den += srow[j]; }
        float rden = 1.0f / den;
        bf16* orow = sb + (seq * 5 + qi) * LDX + head * 32;
        for (int d = 0; d < 32; d++) {
          float o = 0.f;
#pragma unroll
          for (int j = 0; j < 5; j++) o += srow[j] * (float)hb[(seq * 5 + j) * HQ + 64 + d];
          orow[d] = (bf16)(o * rden);
        }
      }
      __syncthreads();
    }

    // Phase B: out_proj + residual + LN1
    {
      const bf16* OW = p.ow[layer]; const float* OB = p.ob[layer];
      v8f acc[6];
#pragma unroll
      for (int i = 0; i < 6; i++) {
        int t = wave + i * 8;
        int mt = t / 16, nt = t % 16;
        acc[i] = wmma_k(sb, LDX, mt * 16, OW, 256, nt * 16, 256, vzero8());
      }
#pragma unroll
      for (int i = 0; i < 6; i++) {             // element-exclusive residual update
        int t = wave + i * 8;
        int mt = t / 16, nt = t % 16;
        int rbase = mt * 16 + ((lane >> 4) << 3);
        int col = nt * 16 + (lane & 15);
        float bias = OB[col];
#pragma unroll
        for (int r = 0; r < 8; r++) {
          int row = rbase + r;
          float v = acc[i][r] + bias + (float)xb[row * LDX + col];
          xb[row * LDX + col] = (bf16)v;
        }
      }
      __syncthreads();
      ln_rows_bf16(xb, LDX, TOK, DM, p.lg[layer], p.lb[layer], false);
      __syncthreads();
    }

    // Phase C: FFN with 64-wide hidden chunking, y accumulated in registers
    {
      const bf16* W1 = p.fw1[layer]; const float* B1 = p.fb1[layer];
      const bf16* W2 = p.fw2[layer]; const float* B2 = p.fb2[layer];
      v8f yacc[6];
#pragma unroll
      for (int i = 0; i < 6; i++) yacc[i] = vzero8();
      for (int c = 0; c < 16; c++) {
#pragma unroll
        for (int i = 0; i < 2; i++) {           // hidden chunk: 3 M x 4 N tiles
          int t = wave + i * 8;
          if (t < 12) {
            int mt = t / 4, nt = t % 4;
            int n0 = c * 64 + nt * 16;
            v8f acc = wmma_k(xb, LDX, mt * 16, W1, 256, n0, 256, vzero8());
            store_tile_bf16_bias(hb, HF, mt * 16, nt * 16, acc, B1, n0, true);
          }
        }
        __syncthreads();
#pragma unroll
        for (int i = 0; i < 6; i++) {           // accumulate y += h_chunk @ W2.T chunk
          int t = wave + i * 8;
          int mt = t / 16, nt = t % 16;
#pragma unroll
          for (int kk = 0; kk < 2; kk++) {
            v16bf a = load_a_lds(hb, HF, mt * 16, kk * 32);
            v16bf b = load_b_glb(W2, 1024, nt * 16, c * 64 + kk * 32);
            yacc[i] = __builtin_amdgcn_wmma_f32_16x16x32_bf16(false, a, false, b,
                                                              (short)0, yacc[i], false, false);
          }
        }
        __syncthreads();
      }
#pragma unroll
      for (int i = 0; i < 6; i++) {
        int t = wave + i * 8;
        int mt = t / 16, nt = t % 16;
        int rbase = mt * 16 + ((lane >> 4) << 3);
        int col = nt * 16 + (lane & 15);
        float bias = B2[col];
#pragma unroll
        for (int r = 0; r < 8; r++) {
          int row = rbase + r;
          float v = yacc[i][r] + bias + (float)xb[row * LDX + col];
          xb[row * LDX + col] = (bf16)v;
        }
      }
      __syncthreads();
      ln_rows_bf16(xb, LDX, TOK, DM, p.ng[layer], p.nb[layer], false);
      __syncthreads();
    }
  }

  // ---- Phase D: gating (cc @ gw1.T -> LN -> gelu -> @ gw2.T -> softmax) ----
  {
#pragma unroll
    for (int i = 0; i < 2; i++) {               // 16 N-tiles over 8 waves
      int nt = wave + i * 8;
      v8f acc = vzero8();
      for (int k0 = 0; k0 < 1280; k0 += 32) {
        v16bf a = load_a_cc(xb, k0);
        v16bf b = load_b_glb(p.gw1, 1280, nt * 16, k0);
        acc = __builtin_amdgcn_wmma_f32_16x16x32_bf16(false, a, false, b,
                                                      (short)0, acc, false, false);
      }
      int rbase = (lane >> 4) << 3;
      int col = nt * 16 + (lane & 15);
      float bias = p.gb1[col];
#pragma unroll
      for (int r = 0; r < 8; r++) gf[(rbase + r) * GLD + col] = acc[r] + bias;
    }
    __syncthreads();
    if (wave < SEQ_PER_BLK) {                   // LN + gelu, f32, wave per row
      float* row = gf + wave * GLD;
      float s = 0.f, s2 = 0.f;
      for (int d = lane; d < DM; d += 32) { float v = row[d]; s += v; s2 += v * v; }
      s = wred(s); s2 = wred(s2);
      float m = s / DM, var = s2 / DM - m * m, inv = rsqrtf(var + 1e-5f);
      for (int d = lane; d < DM; d += 32) {
        float v = (row[d] - m) * inv * p.glg[d] + p.glb[d];
        row[d] = gelu_f(v);
      }
    }
    __syncthreads();
    if (tid < SEQ_PER_BLK) {                    // 5-way logits + softmax per sequence
      int s = tid;
      float logits[5];
      float mx = -1e30f;
      for (int n = 0; n < 5; n++) {
        float a = p.gb2[n];
        const float* w = p.gw2 + n * DM;
        for (int d = 0; d < DM; d++) a += gf[s * GLD + d] * w[d];
        logits[n] = a; mx = fmaxf(mx, a);
      }
      float den = 0.f;
      for (int n = 0; n < 5; n++) { logits[n] = __expf(logits[n] - mx); den += logits[n]; }
      for (int n = 0; n < 5; n++) imp[s * 5 + n] = logits[n] / den;
    }
    __syncthreads();
    for (int idx = tid; idx < TOK * DM; idx += 256) {  // w = fs * imp (scale x in place)
      int tok = idx >> 8, d = idx & 255;
      xb[tok * LDX + d] = (bf16)((float)xb[tok * LDX + d] * imp[tok]);
    }
    __syncthreads();
  }

  // ---- Phase E: projection (w @ pw1.T -> LN -> gelu -> @ pw2.T -> LN -> out) ----
  {
#pragma unroll
    for (int i = 0; i < 4; i++) {               // y1: 32 N-tiles over 8 waves
      int nt = wave + i * 8;
      v8f acc = vzero8();
      for (int k0 = 0; k0 < 1280; k0 += 32) {
        v16bf a = load_a_cc(xb, k0);
        v16bf b = load_b_glb(p.pw1, 1280, nt * 16, k0);
        acc = __builtin_amdgcn_wmma_f32_16x16x32_bf16(false, a, false, b,
                                                      (short)0, acc, false, false);
      }
      store_tile_bf16_bias(h2, H2LD, 0, nt * 16, acc, p.pb1, nt * 16, false);
    }
    __syncthreads();
    ln_rows_bf16(h2, H2LD, SEQ_PER_BLK, 512, p.pl1g, p.pl1b, true);
    __syncthreads();
#pragma unroll
    for (int i = 0; i < 2; i++) {               // y2: 16 N-tiles over 8 waves
      int nt = wave + i * 8;
      v8f acc = wmma_k(h2, H2LD, 0, p.pw2, 512, nt * 16, 512, vzero8());
      int rbase = (lane >> 4) << 3;
      int col = nt * 16 + (lane & 15);
      float bias = p.pb2[col];
#pragma unroll
      for (int r = 0; r < 8; r++) yf[(rbase + r) * GLD + col] = acc[r] + bias;
    }
    __syncthreads();
    if (wave < SEQ_PER_BLK) {                   // final LN, write straight to HBM
      float* row = yf + wave * GLD;
      float s = 0.f, s2 = 0.f;
      for (int d = lane; d < DM; d += 32) { float v = row[d]; s += v; s2 += v * v; }
      s = wred(s); s2 = wred(s2);
      float m = s / DM, var = s2 / DM - m * m, inv = rsqrtf(var + 1e-5f);
      float* orow = p.out + (size_t)(base + wave) * DM;
      for (int d = lane; d < DM; d += 32) {
        orow[d] = (row[d] - m) * inv * p.pl2g[d] + p.pl2b[d];
      }
    }
  }
}

// ---------------- aux kernels ----------------

__global__ void convert_bf16_kernel(const float* s, bf16* d, int n) {
  for (int i = blockIdx.x * blockDim.x + threadIdx.x; i < n; i += gridDim.x * blockDim.x)
    d[i] = (bf16)s[i];
}

__global__ void mask_or_kernel(const unsigned char* m0, const unsigned char* m1,
                               const unsigned char* m2, const unsigned char* m3,
                               const unsigned char* m4, float* o, int n) {
  int i = blockIdx.x * blockDim.x + threadIdx.x;
  if (i < n) o[i] = (m0[i] | m1[i] | m2[i] | m3[i] | m4[i]) ? 1.0f : 0.0f;
}

// ---------------- host launcher ----------------

extern "C" void kernel_launch(void* const* d_in, const int* in_sizes, int n_in,
                              void* d_out, int out_size, void* d_ws, size_t ws_size,
                              hipStream_t stream) {
  (void)in_sizes; (void)n_in; (void)out_size; (void)ws_size;
  bf16* ws = (bf16*)d_ws;
  size_t off = 0;
  auto conv = [&](int idx, size_t n) -> const bf16* {
    bf16* dst = ws + off;
    convert_bf16_kernel<<<512, 256, 0, stream>>>((const float*)d_in[idx], dst, (int)n);
    off += n;
    return dst;
  };

  Params p;
  p.h0 = (const float*)d_in[0];
  p.h1 = (const float*)d_in[2];
  p.h2s = (const float*)d_in[4];
  p.h3 = (const float*)d_in[6];
  p.h4 = (const float*)d_in[8];
  p.type_emb = (const float*)d_in[10];

  const int aw_i[2] = {11, 23}, ab_i[2] = {12, 24}, ow_i[2] = {13, 25}, ob_i[2] = {14, 26};
  const int lg_i[2] = {15, 27}, lb_i[2] = {16, 28}, f1_i[2] = {17, 29}, fb1_i[2] = {18, 30};
  const int f2_i[2] = {19, 31}, fb2_i[2] = {20, 32}, ng_i[2] = {21, 33}, nb_i[2] = {22, 34};

  for (int l = 0; l < 2; l++) {
    p.aw[l]  = conv(aw_i[l], 768 * 256);
    p.ow[l]  = conv(ow_i[l], 256 * 256);
    p.fw1[l] = conv(f1_i[l], 1024 * 256);
    p.fw2[l] = conv(f2_i[l], 256 * 1024);
    p.ab[l]  = (const float*)d_in[ab_i[l]];
    p.ob[l]  = (const float*)d_in[ob_i[l]];
    p.lg[l]  = (const float*)d_in[lg_i[l]];
    p.lb[l]  = (const float*)d_in[lb_i[l]];
    p.fb1[l] = (const float*)d_in[fb1_i[l]];
    p.fb2[l] = (const float*)d_in[fb2_i[l]];
    p.ng[l]  = (const float*)d_in[ng_i[l]];
    p.nb[l]  = (const float*)d_in[nb_i[l]];
  }
  p.gw1 = conv(35, 256 * 1280);
  p.pw1 = conv(41, 512 * 1280);
  p.pw2 = conv(45, 256 * 512);
  p.gb1 = (const float*)d_in[36];
  p.glg = (const float*)d_in[37];
  p.glb = (const float*)d_in[38];
  p.gw2 = (const float*)d_in[39];
  p.gb2 = (const float*)d_in[40];
  p.pb1 = (const float*)d_in[42];
  p.pl1g = (const float*)d_in[43];
  p.pl1b = (const float*)d_in[44];
  p.pb2 = (const float*)d_in[46];
  p.pl2g = (const float*)d_in[47];
  p.pl2b = (const float*)d_in[48];
  p.out = (float*)d_out;

  fusion_kernel<<<NSEQ_TOTAL / SEQ_PER_BLK, 256, 0, stream>>>(p);

  float* mask_out = (float*)d_out + (size_t)NSEQ_TOTAL * DM;
  mask_or_kernel<<<(NSEQ_TOTAL + 255) / 256, 256, 0, stream>>>(
      (const unsigned char*)d_in[1], (const unsigned char*)d_in[3],
      (const unsigned char*)d_in[5], (const unsigned char*)d_in[7],
      (const unsigned char*)d_in[9], mask_out, NSEQ_TOTAL);
}
